// NonLocalConvBlock_33371895890640
// MI455X (gfx1250) — matrix operand order repaired and checked by reading, not compiled, not run on hardware
//
#include <hip/hip_runtime.h>
#include <math.h>

typedef __attribute__((ext_vector_type(16))) __bf16 v16bf;
typedef __attribute__((ext_vector_type(8)))  __bf16 v8bf;
typedef __attribute__((ext_vector_type(8)))  float  v8f;
typedef __attribute__((ext_vector_type(8)))  unsigned int v8u;
typedef __attribute__((ext_vector_type(4)))  float  v4f;

#define N_PIX 9216
#define ICH   32
#define CCH   64
#define KHALF 4608   // keys per wave (2-way key split)

#define CAT16(lo, hi) __builtin_shufflevector((lo), (hi), 0,1,2,3,4,5,6,7,8,9,10,11,12,13,14,15)

__device__ __forceinline__ unsigned int f2bf(float f) {
  unsigned int u = __builtin_bit_cast(unsigned int, f);
  u += 0x7fffu + ((u >> 16) & 1u);           // round-to-nearest-even
  return u >> 16;
}
__device__ __forceinline__ unsigned int pack_bf2(float lo, float hi) {
  return f2bf(lo) | (f2bf(hi) << 16);
}

// ---------------------------------------------------------------------------
// Kernel 1: theta/phi/g projections (fp32 math), bf16 outputs in WMMA layouts:
//   Tb (Q)  channel-major flat  -> rows of the reinterpreted [N,32] Q matrix
//   Pk (K)  pixel-major         -> contiguous key rows for the S^T A-operand
//   Gt (V)  transposed [ch][m]  -> contiguous channel rows for the V^T A-operand
// ---------------------------------------------------------------------------
__global__ void nlb_proj_kernel(const float* __restrict__ x,
                                const float* __restrict__ Wt, const float* __restrict__ bt,
                                const float* __restrict__ Wp, const float* __restrict__ bp,
                                const float* __restrict__ Wg, const float* __restrict__ bg,
                                unsigned short* __restrict__ Tb,
                                unsigned short* __restrict__ Pk,
                                unsigned short* __restrict__ Gt) {
  int p = blockIdx.x * blockDim.x + threadIdx.x;
  if (p >= N_PIX) return;
  float xv[CCH];
#pragma unroll
  for (int c = 0; c < CCH; ++c) xv[c] = x[c * N_PIX + p];

#pragma unroll 4
  for (int j = 0; j < ICH; ++j) {
    float ta = bt[j], pa = bp[j], ga = bg[j];
#pragma unroll
    for (int c = 0; c < CCH; ++c) {
      float xc = xv[c];
      ta = fmaf(Wt[j * CCH + c], xc, ta);
      pa = fmaf(Wp[j * CCH + c], xc, pa);
      ga = fmaf(Wg[j * CCH + c], xc, ga);
    }
    Tb[j * N_PIX + p] = (unsigned short)f2bf(ta);
    Pk[p * ICH + j]   = (unsigned short)f2bf(pa);
    // Gt[jcol][m] = Gflat[32m + jcol]; for f=i*9216+p: jcol=p%32, m=i*288+p/32
    Gt[(p & 31) * N_PIX + j * 288 + (p >> 5)] = (unsigned short)f2bf(ga);
  }
}

// ---------------------------------------------------------------------------
// Online-softmax update for one 16-query set over one 32-key chunk.
// st0/st1 hold, per lane: query = lane&15, keys {8g+r} and {16+8g+r}.
// Returns the P B-fragment (32 keys x 16 queries) for the PV WMMA.
// ---------------------------------------------------------------------------
__device__ __forceinline__ v16bf softmax_update(v8f st0, v8f st1,
                                                float& mrun, float& lrun,
                                                v8f& yt0, v8f& yt1, int grp) {
  float cm = st0[0];
#pragma unroll
  for (int r = 1; r < 8; ++r) cm = fmaxf(cm, st0[r]);
#pragma unroll
  for (int r = 0; r < 8; ++r) cm = fmaxf(cm, st1[r]);
  cm = fmaxf(cm, __shfl_xor(cm, 16));

  float nm = fmaxf(mrun, cm);
  float sc = __expf(mrun - nm);
  mrun = nm;

  float p0[8], p1[8];
  float ps = 0.f;
#pragma unroll
  for (int r = 0; r < 8; ++r) {
    p0[r] = __expf(st0[r] - nm); ps += p0[r];
    p1[r] = __expf(st1[r] - nm); ps += p1[r];
  }
  ps += __shfl_xor(ps, 16);
  lrun = lrun * sc + ps;

#pragma unroll
  for (int r = 0; r < 8; ++r) { yt0[r] = yt0[r] * sc; yt1[r] = yt1[r] * sc; }

  // B-fragment needs keys 16g..16g+15 in this lane: swap halves of the packs.
  unsigned int bw[8];
#pragma unroll
  for (int i = 0; i < 4; ++i) {
    unsigned int own0 = pack_bf2(p0[2 * i], p0[2 * i + 1]);  // keys 8g+2i,+1
    unsigned int own1 = pack_bf2(p1[2 * i], p1[2 * i + 1]);  // keys 16+8g+2i,+1
    unsigned int oth0 = __shfl_xor(own0, 16);
    unsigned int oth1 = __shfl_xor(own1, 16);
    bw[i]     = grp ? oth1 : own0;   // elements 0-7  : keys 16g+0..7
    bw[i + 4] = grp ? own1 : oth0;   // elements 8-15 : keys 16g+8..15
  }
  v8u bwv;
#pragma unroll
  for (int i = 0; i < 8; ++i) bwv[i] = bw[i];
  return __builtin_bit_cast(v16bf, bwv);
}

// ---------------------------------------------------------------------------
// Kernel 2: fused flash attention, transposed dataflow, 32 queries per wave,
// 2-way key split (flash-decoding). 288 query blocks x 2 halves = 576 waves.
// Per 32-key iter: 8 WMMA, 8 b128 loads; K/V fragments reused across the two
// query sets -> half the L2 ingress of a 16-query tile.
// ---------------------------------------------------------------------------
__global__ void __launch_bounds__(32)
nlb_attn_kernel(const unsigned short* __restrict__ Tb,
                const unsigned short* __restrict__ Pk,
                const unsigned short* __restrict__ Gt,
                float* __restrict__ Yp,   // [2][N_PIX][ICH] raw Y^T partials
                float* __restrict__ Mp,   // [2][N_PIX] running max
                float* __restrict__ Lp) { // [2][N_PIX] running sum
  const int lane  = threadIdx.x;
  const int grp   = lane >> 4;
  const int col   = lane & 15;
  const int half  = blockIdx.x & 1;
  const int qbase = (blockIdx.x >> 1) * 32;
  const int kstart = half * KHALF;

  // ---- Q^T B-fragments (loop invariant), query sets A (q) and B (q+16)
  const v8bf* qa = (const v8bf*)(Tb + (size_t)(qbase + col) * ICH + grp * 16);
  const v8bf* qb = (const v8bf*)(Tb + (size_t)(qbase + 16 + col) * ICH + grp * 16);
  const v16bf bQa = CAT16(qa[-grp], qa[2 - grp]);
  const v16bf bQb = CAT16(qb[-grp], qb[2 - grp]);

  v8f ytA0 = {0,0,0,0,0,0,0,0}, ytA1 = {0,0,0,0,0,0,0,0};
  v8f ytB0 = {0,0,0,0,0,0,0,0}, ytB1 = {0,0,0,0,0,0,0,0};
  float mA = -INFINITY, lA = 0.f;
  float mB = -INFINITY, lB = 0.f;
  const v8f cz = {0,0,0,0,0,0,0,0};

  for (int kb = kstart; kb < kstart + KHALF; kb += 32) {
    // ---- K A-fragments (shared by both query sets)
    const unsigned short* k0p = Pk + (size_t)(kb + col) * ICH;
    const unsigned short* k1p = Pk + (size_t)(kb + 16 + col) * ICH;
    v16bf aK0 = CAT16(((const v8bf*)(k0p + grp * 8))[0], ((const v8bf*)(k0p + 16 + grp * 8))[0]);
    v16bf aK1 = CAT16(((const v8bf*)(k1p + grp * 8))[0], ((const v8bf*)(k1p + 16 + grp * 8))[0]);

    v8f stA0 = __builtin_amdgcn_wmma_f32_16x16x32_bf16(false, aK0, false, bQa, (short)0, cz, false, false);
    v8f stA1 = __builtin_amdgcn_wmma_f32_16x16x32_bf16(false, aK1, false, bQa, (short)0, cz, false, false);
    v8f stB0 = __builtin_amdgcn_wmma_f32_16x16x32_bf16(false, aK0, false, bQb, (short)0, cz, false, false);
    v8f stB1 = __builtin_amdgcn_wmma_f32_16x16x32_bf16(false, aK1, false, bQb, (short)0, cz, false, false);

    v16bf bPA = softmax_update(stA0, stA1, mA, lA, ytA0, ytA1, grp);
    v16bf bPB = softmax_update(stB0, stB1, mB, lB, ytB0, ytB1, grp);

    // ---- V^T A-fragments (shared by both query sets)
    const unsigned short* v0p = Gt + (size_t)col * N_PIX + kb;
    const unsigned short* v1p = Gt + (size_t)(16 + col) * N_PIX + kb;
    v16bf aV0 = CAT16(((const v8bf*)(v0p + grp * 16))[0], ((const v8bf*)(v0p + grp * 16 + 8))[0]);
    v16bf aV1 = CAT16(((const v8bf*)(v1p + grp * 16))[0], ((const v8bf*)(v1p + grp * 16 + 8))[0]);

    ytA0 = __builtin_amdgcn_wmma_f32_16x16x32_bf16(false, aV0, false, bPA, (short)0, ytA0, false, false);
    ytA1 = __builtin_amdgcn_wmma_f32_16x16x32_bf16(false, aV1, false, bPA, (short)0, ytA1, false, false);
    ytB0 = __builtin_amdgcn_wmma_f32_16x16x32_bf16(false, aV0, false, bPB, (short)0, ytB0, false, false);
    ytB1 = __builtin_amdgcn_wmma_f32_16x16x32_bf16(false, aV1, false, bPB, (short)0, ytB1, false, false);
  }

  // ---- write raw partials; lane holds ch {8g..8g+7} (yt*0), {16+8g..} (yt*1)
  float* baseA = Yp + ((size_t)half * N_PIX + qbase + col) * ICH;
  *(v4f*)(baseA + 8 * grp)          = (v4f){ytA0[0], ytA0[1], ytA0[2], ytA0[3]};
  *(v4f*)(baseA + 8 * grp + 4)      = (v4f){ytA0[4], ytA0[5], ytA0[6], ytA0[7]};
  *(v4f*)(baseA + 16 + 8 * grp)     = (v4f){ytA1[0], ytA1[1], ytA1[2], ytA1[3]};
  *(v4f*)(baseA + 16 + 8 * grp + 4) = (v4f){ytA1[4], ytA1[5], ytA1[6], ytA1[7]};

  float* baseB = Yp + ((size_t)half * N_PIX + qbase + 16 + col) * ICH;
  *(v4f*)(baseB + 8 * grp)          = (v4f){ytB0[0], ytB0[1], ytB0[2], ytB0[3]};
  *(v4f*)(baseB + 8 * grp + 4)      = (v4f){ytB0[4], ytB0[5], ytB0[6], ytB0[7]};
  *(v4f*)(baseB + 16 + 8 * grp)     = (v4f){ytB1[0], ytB1[1], ytB1[2], ytB1[3]};
  *(v4f*)(baseB + 16 + 8 * grp + 4) = (v4f){ytB1[4], ytB1[5], ytB1[6], ytB1[7]};

  if (grp == 0) {   // one lane per query writes the stats
    Mp[half * N_PIX + qbase + col]      = mA;
    Lp[half * N_PIX + qbase + col]      = lA;
    Mp[half * N_PIX + qbase + 16 + col] = mB;
    Lp[half * N_PIX + qbase + 16 + col] = lB;
  }
}

// ---------------------------------------------------------------------------
// Kernel 2b: log-sum-exp merge of the two key halves + channel softmax.
// ---------------------------------------------------------------------------
__global__ void nlb_merge_kernel(const float* __restrict__ Yp,
                                 const float* __restrict__ Mp,
                                 const float* __restrict__ Lp,
                                 float* __restrict__ Yf) {
  int q = blockIdx.x * blockDim.x + threadIdx.x;
  if (q >= N_PIX) return;
  float m0 = Mp[q], m1 = Mp[N_PIX + q];
  float l0 = Lp[q], l1 = Lp[N_PIX + q];
  float mm = fmaxf(m0, m1);
  float w0 = __expf(m0 - mm), w1 = __expf(m1 - mm);
  float inv = 1.f / (w0 * l0 + w1 * l1);

  const float* y0 = Yp + (size_t)q * ICH;
  const float* y1 = Yp + (size_t)(N_PIX + q) * ICH;
  float yv[ICH];
  float cmx = -INFINITY;
#pragma unroll
  for (int ch = 0; ch < ICH; ++ch) {
    float v = (w0 * y0[ch] + w1 * y1[ch]) * inv;
    yv[ch] = v;
    cmx = fmaxf(cmx, v);
  }
  float cs = 0.f;
#pragma unroll
  for (int ch = 0; ch < ICH; ++ch) {
    yv[ch] = __expf(yv[ch] - cmx);
    cs += yv[ch];
  }
  float ics = 1.f / cs;
  float* dst = Yf + (size_t)q * ICH;   // y_flat[32n + ch]
#pragma unroll
  for (int ch = 0; ch < ICH; ++ch) dst[ch] = yv[ch] * ics;
}

// ---------------------------------------------------------------------------
// Kernel 3: out = Wo * y + bo + x   (y read back channel-major: Yf[i*9216+p])
// ---------------------------------------------------------------------------
__global__ void nlb_out_kernel(const float* __restrict__ x,
                               const float* __restrict__ Wo,
                               const float* __restrict__ bo,
                               const float* __restrict__ Yf,
                               float* __restrict__ out) {
  int p = blockIdx.x * blockDim.x + threadIdx.x;
  if (p >= N_PIX) return;
  float yv[ICH];
#pragma unroll
  for (int i = 0; i < ICH; ++i) yv[i] = Yf[i * N_PIX + p];

#pragma unroll 4
  for (int o = 0; o < CCH; ++o) {
    float acc = bo[o] + x[o * N_PIX + p];
#pragma unroll
    for (int i = 0; i < ICH; ++i) acc = fmaf(Wo[o * ICH + i], yv[i], acc);
    out[o * N_PIX + p] = acc;
  }
}

// ---------------------------------------------------------------------------
extern "C" void kernel_launch(void* const* d_in, const int* in_sizes, int n_in,
                              void* d_out, int out_size, void* d_ws, size_t ws_size,
                              hipStream_t stream) {
  (void)in_sizes; (void)n_in; (void)out_size; (void)ws_size;
  const float* x  = (const float*)d_in[0];
  const float* Wt = (const float*)d_in[1];
  const float* bt = (const float*)d_in[2];
  const float* Wp = (const float*)d_in[3];
  const float* bp = (const float*)d_in[4];
  const float* Wg = (const float*)d_in[5];
  const float* bg = (const float*)d_in[6];
  const float* Wo = (const float*)d_in[7];
  const float* bo = (const float*)d_in[8];
  float* out = (float*)d_out;

  const size_t NE = (size_t)N_PIX * ICH;   // 294912 elements per projection
  unsigned short* Tb = (unsigned short*)d_ws;
  unsigned short* Pk = Tb + NE;
  unsigned short* Gt = Pk + NE;
  float* Yf = (float*)(Gt + NE);           // NE floats
  float* Yp = Yf + NE;                     // 2*NE floats (partials)
  float* Mp = Yp + 2 * NE;                 // 2*N_PIX floats
  float* Lp = Mp + 2 * N_PIX;              // 2*N_PIX floats
  // total workspace: ~5.3 MB

  nlb_proj_kernel<<<36, 256, 0, stream>>>(x, Wt, bt, Wp, bp, Wg, bg, Tb, Pk, Gt);
  nlb_attn_kernel<<<576, 32, 0, stream>>>(Tb, Pk, Gt, Yp, Mp, Lp);
  nlb_merge_kernel<<<36, 256, 0, stream>>>(Yp, Mp, Lp, Yf);
  nlb_out_kernel<<<36, 256, 0, stream>>>(x, Wo, bo, Yf, out);
}
